// CounterfactualDiffusion_36292473651877
// MI455X (gfx1250) — matrix operand order repaired and checked by reading, not compiled, not run on hardware
//
#include <hip/hip_runtime.h>
#include <hip/hip_bf16.h>

// ---------------------------------------------------------------------------
// CounterfactualDiffusion on MI455X (gfx1250), wave32 + v_wmma_f32_16x16x32_bf16
//
//  * persistent per-block 64-row tile through all 50 steps; single in-place
//    LDS activation buffer; fp32 x + bf16 x-mirror for WMMA consumption.
//  * 512 threads / 16 waves: per-wave accum panel = 4 row-groups x 2 n-tiles
//    (64 VGPRs) so B double-buffering fits the flat 256-VGPR file (no spills).
//  * each B-fragment feeds 4 WMMAs (row-group reuse); B is software-pipelined.
//  * weights pre-packed to bf16 in WMMA B-fragment lane order, condition
//    pre-packed to A-fragment order (contiguous 32B/lane loads).
//  * t_emb/target rows of W1 folded into per-step effective bias.
//  * 5-op sigmoid-GELU via v_exp_f32 + v_rcp_f32.
// ---------------------------------------------------------------------------

typedef __attribute__((ext_vector_type(16))) __bf16 v16bf;
typedef __attribute__((ext_vector_type(8)))  __bf16 v8bf;
typedef __attribute__((ext_vector_type(4)))  __bf16 v4bf;
typedef __attribute__((ext_vector_type(8)))  float  v8f;
typedef __attribute__((ext_vector_type(4)))  float  v4f;

#define LATENT 128
#define HIDDEN 512
#define NSTEPS 50
#define BATCH  8192
#define MTILE  64            // rows per block
#define NBLK   (BATCH / MTILE)
#define NTHR   512           // 16 waves
#define HPAD   520           // bf16 stride of hidden LDS rows
#define XPAD   132           // f32 stride of x/np LDS rows
#define HPADX  136           // bf16 stride of x-mirror LDS rows

// workspace layout, offsets in __bf16 elements
#define W1P_OFF   0u        // ksteps=8 (K=256: x||cond), NT=32 -> 131072
#define W2P_OFF   131072u   // ksteps=16, NT=32 -> 262144
#define W3P_OFF   393216u
#define W4P_OFF   655360u   // ksteps=16, NT=8 -> 65536
#define CONDP_OFF 721920u   // 512 rg * 4 ks * 512 = 1048576

static __device__ __forceinline__ v8f wmma_bf16(v16bf a, v16bf b, v8f c) {
  return __builtin_amdgcn_wmma_f32_16x16x32_bf16(false, a, false, b, (short)0, c,
                                                 false, false);
}

// A-fragment gather from bf16 LDS: lane holds row m = lane%16,
// K elements {klo..klo+7} U {16+klo..16+klo+7}, klo = 8*(lane/16).
static __device__ __forceinline__ v16bf load_a_frag(const __bf16* src, int stride,
                                                    int m, int klo, int ks) {
  const __bf16* p = src + m * stride + ks * 32 + klo;
  v8bf lo = *(const v8bf*)p;
  v8bf hi = *(const v8bf*)(p + 16);
  return __builtin_shufflevector(lo, hi, 0, 1, 2, 3, 4, 5, 6, 7, 8, 9, 10, 11,
                                 12, 13, 14, 15);
}

// load 2 B-fragments (this wave's n-tiles) for K-step ks
static __device__ __forceinline__ void loadB2(v16bf b[2],
                                              const __bf16* __restrict__ wp,
                                              int ks, int ntbase, int lane,
                                              int NT) {
#pragma unroll
  for (int i = 0; i < 2; ++i)
    b[i] = *(const v16bf*)(wp + (((unsigned)ks * (unsigned)NT +
                                  (unsigned)(ntbase + i)) * 32u +
                                 (unsigned)lane) * 16u);
}

// sigmoid-form GELU: x * sigmoid(1.702 x) -> 5 VALU ops
static __device__ __forceinline__ float gelu_f(float x) {
  float e = __builtin_amdgcn_exp2f(-2.4554669f * x);  // exp(-1.702x)
  return x * __builtin_amdgcn_rcpf(1.0f + e);
}

// splitmix64 -> Box-Muller, deterministic per (t, element)
static __device__ __forceinline__ float gauss_noise(int t, unsigned idx) {
  unsigned long long z =
      ((unsigned long long)(unsigned)t << 32) | (unsigned long long)idx;
  z += 0x9E3779B97F4A7C15ULL;
  z = (z ^ (z >> 30)) * 0xBF58476D1CE4E5B9ULL;
  z = (z ^ (z >> 27)) * 0x94D049BB133111EBULL;
  z ^= (z >> 31);
  unsigned a = (unsigned)z;
  unsigned b = (unsigned)(z >> 32);
  float u1 = (float)(a >> 8) * 0x1p-24f + 0x1p-25f;
  float u2 = (float)(b >> 8) * 0x1p-24f;
  return sqrtf(-2.0f * __logf(u1)) * __cosf(6.28318530718f * u2);
}

// ---------------------------------------------------------------------------
__global__ void pack_w(const float* __restrict__ W, __bf16* __restrict__ dst,
                       int ksteps, int NT, int N) {
  int tid = blockIdx.x * blockDim.x + threadIdx.x;
  int total = ksteps * NT * 32;
  if (tid >= total) return;
  int lane = tid & 31;
  int frag = tid >> 5;
  int nt = frag % NT;
  int ks = frag / NT;
  int n = nt * 16 + (lane & 15);
  int kbase = ks * 32 + ((lane >> 4) << 3);
  v16bf out;
#pragma unroll
  for (int j = 0; j < 16; ++j) out[j] = (__bf16)W[(kbase + (j < 8 ? j : j + 8)) * N + n];
  *(v16bf*)(dst + (size_t)tid * 16) = out;
}

__global__ void pack_cond(const float* __restrict__ cond,
                          __bf16* __restrict__ dst) {
  int tid = blockIdx.x * blockDim.x + threadIdx.x;
  int total = (BATCH / 16) * 4 * 32;
  if (tid >= total) return;
  int lane = tid & 31;
  int frag = tid >> 5;
  int ks = frag & 3;
  int rg = frag >> 2;
  int m = lane & 15;
  int klo = (lane >> 4) * 8;
  int row = rg * 16 + m;
  v16bf out;
#pragma unroll
  for (int j = 0; j < 16; ++j) {
    int k = ks * 32 + (j < 8 ? (klo + j) : (16 + klo + (j - 8)));
    out[j] = (__bf16)cond[row * LATENT + k];
  }
  *(v16bf*)(dst + (size_t)tid * 16) = out;
}

// hidden->hidden GEMM panel: 4 row-groups x 2 n-tiles, pipelined B.
static __device__ __forceinline__ void hh_compute(const __bf16* hbuf,
                                                  const __bf16* __restrict__ wp,
                                                  const float* bias2, int ntbase,
                                                  int lane, v8f acc[4][2]) {
  const int m0 = lane & 15;
  const int klo = (lane >> 4) * 8;
#pragma unroll
  for (int i = 0; i < 2; ++i)
#pragma unroll
    for (int rg = 0; rg < 4; ++rg)
#pragma unroll
      for (int j = 0; j < 8; ++j) acc[rg][i][j] = bias2[i];

  v16bf b0[2], b1[2];
  loadB2(b0, wp, 0, ntbase, lane, 32);
#pragma unroll 1
  for (int ks = 0; ks < 16; ks += 2) {
    loadB2(b1, wp, ks + 1, ntbase, lane, 32);
#pragma unroll
    for (int rg = 0; rg < 4; ++rg) {
      v16bf a = load_a_frag(hbuf + rg * 16 * HPAD, HPAD, m0, klo, ks);
#pragma unroll
      for (int i = 0; i < 2; ++i) acc[rg][i] = wmma_bf16(a, b0[i], acc[rg][i]);
    }
    loadB2(b0, wp, ks + 2, ntbase, lane, 32);  // tail over-read -> next ws region (safe)
#pragma unroll
    for (int rg = 0; rg < 4; ++rg) {
      v16bf a = load_a_frag(hbuf + rg * 16 * HPAD, HPAD, m0, klo, ks + 1);
#pragma unroll
      for (int i = 0; i < 2; ++i) acc[rg][i] = wmma_bf16(a, b1[i], acc[rg][i]);
    }
  }
}

static __device__ __forceinline__ void hh_store(__bf16* hbuf, int ntbase,
                                                int lane, v8f acc[4][2]) {
  const int ncol = lane & 15;
  const int mhi = (lane >> 4) * 8;
#pragma unroll
  for (int rg = 0; rg < 4; ++rg)
#pragma unroll
    for (int i = 0; i < 2; ++i) {
      int n = (ntbase + i) * 16 + ncol;
#pragma unroll
      for (int j = 0; j < 8; ++j)
        hbuf[(rg * 16 + mhi + j) * HPAD + n] = (__bf16)gelu_f(acc[rg][i][j]);
    }
}

// ---------------------------------------------------------------------------
// Persistent kernel: 128 blocks x 512 threads (16 waves), block = 64 rows.
// ---------------------------------------------------------------------------
__global__ void __launch_bounds__(NTHR)
diff_main(const float* __restrict__ x_init, const float* __restrict__ W1,
          const float* __restrict__ b1, const float* __restrict__ b2,
          const float* __restrict__ b3, const float* __restrict__ b4,
          const int* __restrict__ targp, const __bf16* __restrict__ ws,
          float* __restrict__ out) {
  __shared__ alignas(16) __bf16 hbuf[MTILE * HPAD];  // 66.5 KB (in-place layers)
  __shared__ alignas(16) float  xbuf[MTILE * XPAD];  // 33.8 KB (fp32 x)
  __shared__ alignas(16) __bf16 xbf[MTILE * HPADX];  // 17.4 KB (bf16 x mirror)
  float* npf = (float*)hbuf;  // noise-pred f32 region aliases hbuf (after L4)

  const int tid = threadIdx.x;
  const int lane = tid & 31;
  const int w = tid >> 5;        // wave 0..15
  const int ntbase = w * 2;      // layers 1-3: n-tiles [2w, 2w+1]
  const int blk = blockIdx.x;
  const int m0 = lane & 15;
  const int klo = (lane >> 4) * 8;
  const int mhi = (lane >> 4) * 8;
  const int ncol = lane & 15;

  const __bf16* w1p = ws + W1P_OFF;
  const __bf16* w2p = ws + W2P_OFF;
  const __bf16* w3p = ws + W3P_OFF;
  const __bf16* w4p = ws + W4P_OFF;
  const __bf16* condp = ws + CONDP_OFF;

  // x tile load: thread -> row tid/8, 16-col run (tid%8)*16
  const int xr = tid >> 3;
  const int xc0 = (tid & 7) * 16;
  {
    const float* src = x_init + ((size_t)blk * MTILE + xr) * LATENT + xc0;
#pragma unroll
    for (int i = 0; i < 4; ++i) {
      v4f v = *(const v4f*)(src + i * 4);
      *(v4f*)&xbuf[xr * XPAD + xc0 + i * 4] = v;
      v4bf vb;
#pragma unroll
      for (int q = 0; q < 4; ++q) vb[q] = (__bf16)v[q];
      *(v4bf*)&xbf[xr * HPADX + xc0 + i * 4] = vb;
    }
  }
  const float targ = (*targp != 0) ? 0.0f : 1.0f;

  // hoist t-invariant biases (2 owned n-tiles for L1-L3; L4 tile = w&7)
  float base1[2], w1t[2], bias2[2], bias3[2];
#pragma unroll
  for (int i = 0; i < 2; ++i) {
    int n = (ntbase + i) * 16 + ncol;
    base1[i] = b1[n] + targ * W1[257 * HIDDEN + n];
    w1t[i] = W1[256 * HIDDEN + n];
    bias2[i] = b2[n];
    bias3[i] = b3[n];
  }
  const int nt4 = w & 7;            // layer-4 n-tile
  const int rgb4 = (w >> 3) * 2;    // layer-4 row-group pair base
  const float bias4v = b4[nt4 * 16 + ncol];

  // running alphas_cumprod: acp(t) = prod_{i<=t}(1-beta_i), peeled per step
  float acp = 1.0f;
  for (int i2 = 0; i2 < NSTEPS; ++i2)
    acp *= (1.0f - (1e-4f + (0.02f - 1e-4f) * ((float)i2 / 49.0f)));
  __syncthreads();

  for (int t = NSTEPS - 1; t >= 0; --t) {
    float beta_t = 1e-4f + (0.02f - 1e-4f) * ((float)t / 49.0f);
    float cnp = beta_t * rsqrtf(1.0f - acp);
    float inv_sa = rsqrtf(1.0f - beta_t);
    float sigma = sqrtf(beta_t);
    float tt = (float)t / (float)NSTEPS;
    acp /= (1.0f - beta_t);  // -> prod_{i<=t-1} for next iteration

    // ---- Layer 1: [64 x 256] (x||cond) @ W1[0:256,:] + b1_eff(t) ----
    {
      v8f acc[4][2];
#pragma unroll
      for (int i = 0; i < 2; ++i) {
        float be = base1[i] + tt * w1t[i];
#pragma unroll
        for (int rg = 0; rg < 4; ++rg)
#pragma unroll
          for (int j = 0; j < 8; ++j) acc[rg][i][j] = be;
      }
      v16bf b0[2], b1v[2];
      loadB2(b0, w1p, 0, ntbase, lane, 32);
#pragma unroll
      for (int ks = 0; ks < 8; ks += 2) {
        loadB2(b1v, w1p, ks + 1, ntbase, lane, 32);
#pragma unroll
        for (int rg = 0; rg < 4; ++rg) {
          v16bf a = (ks < 4)
                        ? load_a_frag(xbf + rg * 16 * HPADX, HPADX, m0, klo, ks)
                        : *(const v16bf*)(condp +
                                          (((unsigned)(blk * 4 + rg) * 4u +
                                            (unsigned)(ks - 4)) * 32u +
                                           (unsigned)lane) * 16u);
#pragma unroll
          for (int i = 0; i < 2; ++i) acc[rg][i] = wmma_bf16(a, b0[i], acc[rg][i]);
        }
        loadB2(b0, w1p, ks + 2, ntbase, lane, 32);  // over-read lands in W2P (safe)
#pragma unroll
        for (int rg = 0; rg < 4; ++rg) {
          int kk = ks + 1;
          v16bf a = (kk < 4)
                        ? load_a_frag(xbf + rg * 16 * HPADX, HPADX, m0, klo, kk)
                        : *(const v16bf*)(condp +
                                          (((unsigned)(blk * 4 + rg) * 4u +
                                            (unsigned)(kk - 4)) * 32u +
                                           (unsigned)lane) * 16u);
#pragma unroll
          for (int i = 0; i < 2; ++i) acc[rg][i] = wmma_bf16(a, b1v[i], acc[rg][i]);
        }
      }
      hh_store(hbuf, ntbase, lane, acc);  // first writer of hbuf this step
    }
    __syncthreads();

    // ---- Layer 2 (in-place: read all, sync, write own columns) ----
    {
      v8f acc[4][2];
      hh_compute(hbuf, w2p, bias2, ntbase, lane, acc);
      __syncthreads();
      hh_store(hbuf, ntbase, lane, acc);
    }
    __syncthreads();

    // ---- Layer 3 ----
    {
      v8f acc[4][2];
      hh_compute(hbuf, w3p, bias3, ntbase, lane, acc);
      __syncthreads();
      hh_store(hbuf, ntbase, lane, acc);
    }
    __syncthreads();

    // ---- Layer 4: [64 x 512] @ W4[512 x 128] + b4 ----
    // wave w -> n-tile (w&7), row-groups {rgb4, rgb4+1}
    {
      v8f acc[2];
#pragma unroll
      for (int r = 0; r < 2; ++r)
#pragma unroll
        for (int j = 0; j < 8; ++j) acc[r][j] = bias4v;

      v16bf b0 = *(const v16bf*)(w4p + (((unsigned)nt4) * 32u +
                                        (unsigned)lane) * 16u);
      v16bf b1v;
#pragma unroll 1
      for (int ks = 0; ks < 16; ks += 2) {
        b1v = *(const v16bf*)(w4p + (((unsigned)(ks + 1) * 8u + (unsigned)nt4) * 32u +
                                     (unsigned)lane) * 16u);
#pragma unroll
        for (int r = 0; r < 2; ++r) {
          v16bf a = load_a_frag(hbuf + (rgb4 + r) * 16 * HPAD, HPAD, m0, klo, ks);
          acc[r] = wmma_bf16(a, b0, acc[r]);
        }
        b0 = *(const v16bf*)(w4p + (((unsigned)(ks + 2) * 8u + (unsigned)nt4) * 32u +
                                    (unsigned)lane) * 16u);  // over-read -> condp
#pragma unroll
        for (int r = 0; r < 2; ++r) {
          v16bf a = load_a_frag(hbuf + (rgb4 + r) * 16 * HPAD, HPAD, m0, klo, ks + 1);
          acc[r] = wmma_bf16(a, b1v, acc[r]);
        }
      }
      __syncthreads();  // everyone done reading hbuf before f32 alias write
      int n = nt4 * 16 + ncol;
#pragma unroll
      for (int r = 0; r < 2; ++r)
#pragma unroll
        for (int j = 0; j < 8; ++j)
          npf[((rgb4 + r) * 16 + mhi + j) * XPAD + n] = acc[r][j];
    }
    __syncthreads();

    // ---- x update + noise; refresh fp32 x and bf16 mirror ----
#pragma unroll
    for (int i = 0; i < 4; ++i) {
      v4f xv = *(v4f*)&xbuf[xr * XPAD + xc0 + i * 4];
      v4f nv = *(v4f*)&npf[xr * XPAD + xc0 + i * 4];
      v4bf xb;
#pragma unroll
      for (int q = 0; q < 4; ++q) {
        float noise = 0.0f;
        if (t > 0)
          noise = gauss_noise(
              t, (unsigned)((blk * MTILE + xr) * LATENT + xc0 + i * 4 + q));
        xv[q] = (xv[q] - cnp * nv[q]) * inv_sa + sigma * noise;
        xb[q] = (__bf16)xv[q];
      }
      *(v4f*)&xbuf[xr * XPAD + xc0 + i * 4] = xv;
      *(v4bf*)&xbf[xr * HPADX + xc0 + i * 4] = xb;
    }
    __syncthreads();
  }

  // write x_final (vectorized)
  {
    float* dst = out + ((size_t)blk * MTILE + xr) * LATENT + xc0;
#pragma unroll
    for (int i = 0; i < 4; ++i)
      *(v4f*)(dst + i * 4) = *(const v4f*)&xbuf[xr * XPAD + xc0 + i * 4];
  }
}

// ---------------------------------------------------------------------------
extern "C" void kernel_launch(void* const* d_in, const int* in_sizes, int n_in,
                              void* d_out, int out_size, void* d_ws,
                              size_t ws_size, hipStream_t stream) {
  (void)in_sizes; (void)n_in; (void)out_size; (void)ws_size;
  const float* cond   = (const float*)d_in[0];
  const float* x_init = (const float*)d_in[1];
  const float* W1     = (const float*)d_in[2];
  const float* b1     = (const float*)d_in[3];
  const float* W2     = (const float*)d_in[4];
  const float* b2     = (const float*)d_in[5];
  const float* W3     = (const float*)d_in[6];
  const float* b3     = (const float*)d_in[7];
  const float* W4     = (const float*)d_in[8];
  const float* b4     = (const float*)d_in[9];
  const int*   targ   = (const int*)d_in[10];
  __bf16* ws = (__bf16*)d_ws;
  float* out = (float*)d_out;

  pack_w<<<(8 * 32 * 32) / 256, 256, 0, stream>>>(W1, ws + W1P_OFF, 8, 32, 512);
  pack_w<<<(16 * 32 * 32) / 256, 256, 0, stream>>>(W2, ws + W2P_OFF, 16, 32, 512);
  pack_w<<<(16 * 32 * 32) / 256, 256, 0, stream>>>(W3, ws + W3P_OFF, 16, 32, 512);
  pack_w<<<(16 * 8 * 32) / 256, 256, 0, stream>>>(W4, ws + W4P_OFF, 16, 8, 128);
  pack_cond<<<((BATCH / 16) * 4 * 32) / 256, 256, 0, stream>>>(cond,
                                                               ws + CONDP_OFF);

  diff_main<<<NBLK, NTHR, 0, stream>>>(x_init, W1, b1, b2, b3, b4, targ, ws, out);
}
